// FrameTransformerEncoder_82145544503595
// MI455X (gfx1250) — compile-verified, matrix operand order checked by reading, not run on hardware
//
#include <hip/hip_runtime.h>
#include <math.h>

// ---------------------------------------------------------------------------
// FrameTransformerEncoder for MI455X (gfx1250, wave32).
// All GEMMs (q/k/v/ol pointwise, QK^T, AV) run through one generic batched
// WMMA kernel: one wave computes a 32x32 f32 tile as 2x2 sub-tiles of
// v_wmma_f32_16x16x32_f16 (4 WMMAs per K-step, each fragment reused twice).
// Every intermediate is laid out so that BOTH operands are K-contiguous:
// all fragment loads are b128 vector loads (no scalar gather path at all).
// qk logits (+mask, *1/16) go directly to d_out (part of the returned tuple)
// and are re-read by the softmax kernel.
// ---------------------------------------------------------------------------

typedef __attribute__((ext_vector_type(16))) _Float16 v16h;
typedef __attribute__((ext_vector_type(8)))  float    v8f;

static constexpr int  kC  = 8;
static constexpr int  kM  = 8;     // MAPS
static constexpr int  kF  = 256;
static constexpr int  kNH = 8;
static constexpr int  kD  = 32;
static constexpr int  kW  = 768;
static constexpr long long kFW = (long long)kF * kW;            // 196608
static constexpr long long kT  = (long long)kM * kFW;           // 1572864
static constexpr long long kWW = (long long)kM * kNH * kW * kW; // 37748736

// ---------------------------------------------------------------------------
// Fragment loader. `base` points at this lane's row (A) / column (B); K is
// contiguous. ISA 7.12.2 16-bit layout: half 0 -> K = e (0-7), 16+(e-8);
// half 1 -> K = 8+e, 24+(e-8). Two contiguous 8-float runs -> 4x float4.
// ---------------------------------------------------------------------------
__device__ __forceinline__ v16h ft_load_frag(const float* __restrict__ base,
                                             int k0, int half)
{
    const float4* p0 = reinterpret_cast<const float4*>(base + k0 + 8 * half);
    const float4* p1 = reinterpret_cast<const float4*>(base + k0 + 16 + 8 * half);
    const float4 a0 = p0[0], a1 = p0[1];
    const float4 b0 = p1[0], b1 = p1[1];
    v16h f;
    f[0]  = (_Float16)a0.x; f[1]  = (_Float16)a0.y;
    f[2]  = (_Float16)a0.z; f[3]  = (_Float16)a0.w;
    f[4]  = (_Float16)a1.x; f[5]  = (_Float16)a1.y;
    f[6]  = (_Float16)a1.z; f[7]  = (_Float16)a1.w;
    f[8]  = (_Float16)b0.x; f[9]  = (_Float16)b0.y;
    f[10] = (_Float16)b0.z; f[11] = (_Float16)b0.w;
    f[12] = (_Float16)b1.x; f[13] = (_Float16)b1.y;
    f[14] = (_Float16)b1.z; f[15] = (_Float16)b1.w;
    return f;
}

// ---------------------------------------------------------------------------
// Generic batched GEMM:  C[b] = alpha * A[b] x B[b] (+ bias2d), f16 WMMA path.
//   A elem: A[b*sAb + row*lda_r + k]                 (M x K, k contiguous)
//   B elem: B[b*sBb + col*ldb_n + k]                 (K x N, k contiguous)
//   C elem: C[(b/cInner)*sCb_o + (b%cInner)*sCb_i + row*ldc_r + col*ldc_n]
//   bias2d (optional, batch-invariant): bias2d[row*ldbias + col]
// M,N multiples of 32; K multiple of 32. One wave per 32x32 output tile.
// ---------------------------------------------------------------------------
__global__ void ft_gemm_wmma_f16(
    const float* __restrict__ A, const float* __restrict__ B,
    const float* __restrict__ bias2d, float* __restrict__ C,
    int M, int N, int K,
    long long sAb, int lda_r,
    long long sBb, int ldb_n,
    int cInner, long long sCb_o, long long sCb_i, int ldc_r, int ldc_n,
    int ldbias, float alpha, int nbatch)
{
    const int lane = threadIdx.x & 31;
    const int wave = (int)((blockIdx.x * blockDim.x + threadIdx.x) >> 5);
    const int nt = N >> 5;                 // 32-wide column tiles
    const int mt = M >> 5;                 // 32-tall row tiles
    const int tilesPerBatch = mt * nt;
    if (wave >= tilesPerBatch * nbatch) return;   // wave-uniform exit

    const int b    = wave / tilesPerBatch;
    const int t    = wave - b * tilesPerBatch;
    const int row0 = (t / nt) << 5;
    const int col0 = (t % nt) << 5;

    const float* Ab = A + (long long)b * sAb;
    const float* Bb = B + (long long)b * sBb;
    float*       Cb = C + (long long)(b / cInner) * sCb_o
                        + (long long)(b % cInner) * sCb_i;

    const int half = lane >> 4;
    const int l    = lane & 15;

    const float* arow0 = Ab + (long long)(row0 + l) * lda_r;
    const float* arow1 = arow0 + (long long)16 * lda_r;
    const float* bcol0 = Bb + (long long)(col0 + l) * ldb_n;
    const float* bcol1 = bcol0 + (long long)16 * ldb_n;

    v8f acc00 = {}, acc01 = {}, acc10 = {}, acc11 = {};
    for (int k0 = 0; k0 < K; k0 += 32) {
        const v16h fa0 = ft_load_frag(arow0, k0, half);
        const v16h fa1 = ft_load_frag(arow1, k0, half);
        const v16h fb0 = ft_load_frag(bcol0, k0, half);
        const v16h fb1 = ft_load_frag(bcol1, k0, half);
        acc00 = __builtin_amdgcn_wmma_f32_16x16x32_f16(false, fa0, false, fb0, (short)0, acc00, false, false);
        acc01 = __builtin_amdgcn_wmma_f32_16x16x32_f16(false, fa0, false, fb1, (short)0, acc01, false, false);
        acc10 = __builtin_amdgcn_wmma_f32_16x16x32_f16(false, fa1, false, fb0, (short)0, acc10, false, false);
        acc11 = __builtin_amdgcn_wmma_f32_16x16x32_f16(false, fa1, false, fb1, (short)0, acc11, false, false);
    }

    // Store 4 sub-tiles. C/D layout: VGPR r -> M = r + 8*half, N = lane&15.
#pragma unroll
    for (int i = 0; i < 2; ++i) {
#pragma unroll
        for (int j = 0; j < 2; ++j) {
            const v8f acc = (i == 0) ? (j == 0 ? acc00 : acc01)
                                     : (j == 0 ? acc10 : acc11);
            const int colj = col0 + j * 16 + l;
#pragma unroll
            for (int r = 0; r < 8; ++r) {
                const int row = row0 + i * 16 + r + 8 * half;
                float v = alpha * acc[r];
                if (bias2d) v += bias2d[(long long)row * ldbias + colj];
                Cb[(long long)row * ldc_r + (long long)colj * ldc_n] = v;
            }
        }
    }
}

// ---------------------------------------------------------------------------
// LayerNorm over the feature axis F=256 at each (c, w) position.
// ---------------------------------------------------------------------------
__global__ void ft_layernorm(const float* __restrict__ x,
                             const float* __restrict__ w,
                             const float* __restrict__ b,
                             float* __restrict__ y)
{
    const int pos  = blockIdx.x;          // c*W + wpos
    const int wpos = pos % kW;
    const int c    = pos / kW;
    const int f    = threadIdx.x;

    const long long idx = (long long)c * kFW + (long long)f * kW + wpos;
    const float v = x[idx];

    __shared__ float red[256];
    red[f] = v;
    __syncthreads();
    for (int s = 128; s > 0; s >>= 1) { if (f < s) red[f] += red[f + s]; __syncthreads(); }
    const float mu = red[0] * (1.0f / 256.0f);
    __syncthreads();
    const float d = v - mu;
    red[f] = d * d;
    __syncthreads();
    for (int s = 128; s > 0; s >>= 1) { if (f < s) red[f] += red[f + s]; __syncthreads(); }
    const float var = red[0] * (1.0f / 256.0f);

    y[idx] = d * rsqrtf(var + 1e-8f) * w[c * kF + f] + b[c * kF + f];
}

// ---------------------------------------------------------------------------
// 3x3 conv, zero padding 1, on the (F, W) grid; virtual channel concat of two
// inputs. transposedOut=1 stores (co, w, f) so downstream GEMM B operands are
// K(=f)-contiguous.
// ---------------------------------------------------------------------------
__global__ void ft_conv3x3(const float* __restrict__ in0, int c0,
                           const float* __restrict__ in1, int c1,
                           const float* __restrict__ wgt,
                           const float* __restrict__ bias,
                           float* __restrict__ out, int cout, int transposedOut)
{
    const long long total = (long long)cout * kFW;
    const long long idx = (long long)blockIdx.x * blockDim.x + threadIdx.x;
    if (idx >= total) return;
    const int w  = (int)(idx % kW);
    const int f  = (int)((idx / kW) % kF);
    const int co = (int)(idx / kFW);
    const int cin = c0 + c1;

    float acc = bias[co];
    for (int ci = 0; ci < cin; ++ci) {
        const float* src = (ci < c0) ? (in0 + (long long)ci * kFW)
                                     : (in1 + (long long)(ci - c0) * kFW);
        const float* wp = wgt + (long long)(co * cin + ci) * 9;
#pragma unroll
        for (int kf = -1; kf <= 1; ++kf) {
#pragma unroll
            for (int kw = -1; kw <= 1; ++kw) {
                const int ff = f + kf, ww = w + kw;
                if (ff >= 0 && ff < kF && ww >= 0 && ww < kW)
                    acc += wp[(kf + 1) * 3 + (kw + 1)] * src[(long long)ff * kW + ww];
            }
        }
    }
    const long long oidx = transposedOut
        ? ((long long)co * kFW + (long long)w * kF + f)   // (co, w, f)
        : idx;                                            // (co, f, w)
    out[oidx] = acc;
}

// ---------------------------------------------------------------------------
// heads_split (m,f,w) -> heads tensor, with optional interleaved rotary.
//   dMajorOut=0: out[m,nh,w,d]  (q/k layout: GEMM A operand, row=w, k=d)
//   dMajorOut=1: out[m,nh,d,w]  (v layout:   GEMM B operand, col=d, k=w)
// ---------------------------------------------------------------------------
__global__ void ft_rotary_split(const float* __restrict__ in,
                                float* __restrict__ out,
                                int applyRot, int dMajorOut)
{
    const long long total = (long long)kM * kNH * kW * (kD / 2);
    long long idx = (long long)blockIdx.x * blockDim.x + threadIdx.x;
    if (idx >= total) return;
    const int p  = (int)(idx & (kD / 2 - 1)); idx >>= 4;
    const int w  = (int)(idx % kW);            idx /= kW;
    const int nh = (int)(idx & (kNH - 1));
    const int m  = (int)(idx >> 3);

    const long long inb = (long long)m * kFW + (long long)(nh * kD + 2 * p) * kW + w;
    const float t0 = in[inb];
    const float t1 = in[inb + kW];
    float o0 = t0, o1 = t1;
    if (applyRot) {
        const float invf = __powf(10000.0f, -(float)p / 16.0f);  // 10000^(-(2p)/D)
        const float ang  = (float)w * invf;
        const float s = __sinf(ang), c = __cosf(ang);
        o0 = t0 * c - t1 * s;
        o1 = t1 * c + t0 * s;
    }
    const long long head = (long long)(m * kNH + nh);
    if (dMajorOut) {
        const long long ob = (head * kD + 2 * p) * kW + w;   // (m,nh,d,w)
        out[ob]      = o0;
        out[ob + kW] = o1;
    } else {
        const long long ob = (head * kW + w) * kD + 2 * p;   // (m,nh,w,d)
        out[ob]     = o0;
        out[ob + 1] = o1;
    }
}

// ---------------------------------------------------------------------------
// Row softmax over length 768 (one 256-thread block per row).
// ---------------------------------------------------------------------------
__global__ void ft_softmax768(const float* __restrict__ in, float* __restrict__ out)
{
    const long long row = blockIdx.x;
    const float* r = in  + row * kW;
    float*       o = out + row * kW;
    const int t = threadIdx.x;

    float v0 = r[t], v1 = r[t + 256], v2 = r[t + 512];
    __shared__ float red[256];
    red[t] = fmaxf(v0, fmaxf(v1, v2));
    __syncthreads();
    for (int s = 128; s > 0; s >>= 1) { if (t < s) red[t] = fmaxf(red[t], red[t + s]); __syncthreads(); }
    const float m = red[0];
    __syncthreads();
    v0 = __expf(v0 - m); v1 = __expf(v1 - m); v2 = __expf(v2 - m);
    red[t] = v0 + v1 + v2;
    __syncthreads();
    for (int s = 128; s > 0; s >>= 1) { if (t < s) red[t] += red[t + s]; __syncthreads(); }
    const float inv = 1.0f / red[0];
    o[t] = v0 * inv; o[t + 256] = v1 * inv; o[t + 512] = v2 * inv;
}

// ---------------------------------------------------------------------------
// Depthwise channel mix: out[n,f,w] = sum_m dw[n,m] * in[m,f,w]  (8x8).
// ---------------------------------------------------------------------------
__global__ void ft_chan_mix(const float* __restrict__ dw,
                            const float* __restrict__ in, float* __restrict__ out)
{
    const long long idx = (long long)blockIdx.x * blockDim.x + threadIdx.x;
    if (idx >= kT) return;
    const int n = (int)(idx / kFW);
    const long long fw = idx % kFW;
    float acc = 0.0f;
#pragma unroll
    for (int m = 0; m < kM; ++m) acc += dw[n * kM + m] * in[(long long)m * kFW + fw];
    out[idx] = acc;
}

__global__ void ft_add(const float* __restrict__ a, const float* __restrict__ b,
                       float* __restrict__ out, long long n)
{
    const long long idx = (long long)blockIdx.x * blockDim.x + threadIdx.x;
    if (idx < n) out[idx] = a[idx] + b[idx];
}

__global__ void ft_gelu(float* __restrict__ x, long long n)
{
    const long long idx = (long long)blockIdx.x * blockDim.x + threadIdx.x;
    if (idx < n) {
        const float v = x[idx];
        x[idx] = 0.5f * v * (1.0f + erff(v * 0.70710678118654752f));
    }
}

// ---------------------------------------------------------------------------
// Host side
// ---------------------------------------------------------------------------
namespace {

struct AttnP {
    const float *k_pw, *kc_b, *kc_w, *oc_b, *oc_w, *ol_dw, *ol_pw,
                *q_pw, *qc_b, *qc_w, *v_pw, *vc_b, *vc_w;
};

inline unsigned blocksFor(long long n, int t) { return (unsigned)((n + t - 1) / t); }

inline void launch_gemm(const float* A, const float* B, const float* bias2d, float* C,
                        int M, int N, int K,
                        long long sAb, int lda_r,
                        long long sBb, int ldb_n,
                        int cInner, long long sCb_o, long long sCb_i,
                        int ldc_r, int ldc_n,
                        int ldbias, float alpha, int nbatch, hipStream_t stream)
{
    const long long waves = (long long)(M / 32) * (N / 32) * nbatch;
    const long long thr = waves * 32;
    ft_gemm_wmma_f16<<<blocksFor(thr, 256), 256, 0, stream>>>(
        A, B, bias2d, C, M, N, K,
        sAb, lda_r, sBb, ldb_n, cInner, sCb_o, sCb_i, ldc_r, ldc_n,
        ldbias, alpha, nbatch);
}

// Full attention block: x_arg (normed input), mem (k/v source), qk -> d_out slab.
// conv_t : (m, w, f) transposed conv output
// merged_t: (m, w, f) transposed attention-merge buffer
inline void run_attention(const float* x_arg, const float* mem, const AttnP& p,
                          const float* mask, float* qk_out, float* z_out,
                          float* conv_t, float* pw, float* qh, float* kh, float* vh_t,
                          float* merged_t, float* opw, float* omix, float* probs,
                          hipStream_t stream)
{
    const long long hT = (long long)kM * kNH * kW * (kD / 2);

    // q path: conv (transposed out) -> per-map FxF GEMM -> rotary + head split
    ft_conv3x3<<<blocksFor(kT, 256), 256, 0, stream>>>(x_arg, kC, nullptr, 0, p.qc_w, p.qc_b, conv_t, kM, 1);
    launch_gemm(p.q_pw, conv_t, nullptr, pw, kF, kW, kF,
                (long long)kF * kF, kF,  kFW, kF,
                1, kFW, 0, kW, 1,  0, 1.0f, kM, stream);
    ft_rotary_split<<<blocksFor(hT, 256), 256, 0, stream>>>(pw, qh, 1, 0);

    // k path
    ft_conv3x3<<<blocksFor(kT, 256), 256, 0, stream>>>(mem, kC, nullptr, 0, p.kc_w, p.kc_b, conv_t, kM, 1);
    launch_gemm(p.k_pw, conv_t, nullptr, pw, kF, kW, kF,
                (long long)kF * kF, kF,  kFW, kF,
                1, kFW, 0, kW, 1,  0, 1.0f, kM, stream);
    ft_rotary_split<<<blocksFor(hT, 256), 256, 0, stream>>>(pw, kh, 1, 0);

    // v path (no rotary; d-major output so AV's B operand is K-contiguous)
    ft_conv3x3<<<blocksFor(kT, 256), 256, 0, stream>>>(mem, kC, nullptr, 0, p.vc_w, p.vc_b, conv_t, kM, 1);
    launch_gemm(p.v_pw, conv_t, nullptr, pw, kF, kW, kF,
                (long long)kF * kF, kF,  kFW, kF,
                1, kFW, 0, kW, 1,  0, 1.0f, kM, stream);
    ft_rotary_split<<<blocksFor(hT, 256), 256, 0, stream>>>(pw, vh_t, 0, 1);

    // qk = q k^T / sqrt(F) + mask  -> straight into d_out (also softmax input)
    launch_gemm(qh, kh, mask, qk_out, kW, kW, kD,
                (long long)kW * kD, kD,            // A = q (W x D) row-major
                (long long)kW * kD, kD,            // B = k (col j -> j*D, k contiguous)
                1, (long long)kW * kW, 0, kW, 1,   // C = qk (W x W)
                kW, 1.0f / 16.0f, kM * kNH, stream);

    // softmax over last axis
    ft_softmax768<<<(unsigned)((long long)kM * kNH * kW), 256, 0, stream>>>(qk_out, probs);

    // a = probs @ v  -> merged_t (m, w, f=nh*D+d): batch base m*FW + nh*D
    launch_gemm(probs, vh_t, nullptr, merged_t, kW, kD, kW,
                (long long)kW * kW, kW,            // A = probs (W x W)
                (long long)kW * kD, kW,            // B = v_t (col d -> d*W, k contiguous)
                kNH, kFW, (long long)kD,           // cInner=8: base = m*FW + nh*D
                kF, 1,                             // elem = w*F + d
                0, 1.0f, kM * kNH, stream);

    // ol pointwise (B = merged_t, K-contiguous) + depthwise channel mix
    launch_gemm(p.ol_pw, merged_t, nullptr, opw, kF, kW, kF,
                (long long)kF * kF, kF,  kFW, kF,
                1, kFW, 0, kW, 1,  0, 1.0f, kM, stream);
    ft_chan_mix<<<blocksFor(kT, 256), 256, 0, stream>>>(p.ol_dw, opw, omix);

    // output conv on concat(x_arg, o), regular (c,f,w) output
    ft_conv3x3<<<blocksFor(kT, 256), 256, 0, stream>>>(x_arg, kC, omix, kM, p.oc_w, p.oc_b, z_out, kC, 0);
}

inline AttnP make_attn(void* const* d_in, int base)
{
    AttnP p;
    p.k_pw  = (const float*)d_in[base + 0];
    p.kc_b  = (const float*)d_in[base + 1];
    p.kc_w  = (const float*)d_in[base + 2];
    p.oc_b  = (const float*)d_in[base + 3];
    p.oc_w  = (const float*)d_in[base + 4];
    p.ol_dw = (const float*)d_in[base + 5];
    p.ol_pw = (const float*)d_in[base + 6];
    p.q_pw  = (const float*)d_in[base + 7];
    p.qc_b  = (const float*)d_in[base + 8];
    p.qc_w  = (const float*)d_in[base + 9];
    p.v_pw  = (const float*)d_in[base + 10];
    p.vc_b  = (const float*)d_in[base + 11];
    p.vc_w  = (const float*)d_in[base + 12];
    return p;
}

} // namespace

extern "C" void kernel_launch(void* const* d_in, const int* in_sizes, int n_in,
                              void* d_out, int out_size, void* d_ws, size_t ws_size,
                              hipStream_t stream)
{
    (void)in_sizes; (void)n_in; (void)out_size; (void)ws_size;

    // Input order: x, skip, mask1, mask2, then params flattened as jax tree
    // leaves (dicts in sorted-key order):
    //   attn1[13], attn2[13], c1_b, c1_w, c2_b, c2_w, n1_b, n1_w, n2_b, n2_w, n3_b, n3_w
    const float* X    = (const float*)d_in[0];
    const float* SKIP = (const float*)d_in[1];
    const float* M1   = (const float*)d_in[2];
    const float* M2   = (const float*)d_in[3];
    const AttnP  A1 = make_attn(d_in, 4);
    const AttnP  A2 = make_attn(d_in, 17);
    const float* c1_b = (const float*)d_in[30];
    const float* c1_w = (const float*)d_in[31];
    const float* c2_b = (const float*)d_in[32];
    const float* c2_w = (const float*)d_in[33];
    const float* n1_b = (const float*)d_in[34];
    const float* n1_w = (const float*)d_in[35];
    const float* n2_b = (const float*)d_in[36];
    const float* n2_w = (const float*)d_in[37];
    const float* n3_b = (const float*)d_in[38];
    const float* n3_w = (const float*)d_in[39];

    // Output tuple: h (T) | qk1 (WW) | qk2 (WW)
    float* out_h   = (float*)d_out;
    float* out_qk1 = out_h + kT;
    float* out_qk2 = out_qk1 + kWW;

    // Workspace carve-up (floats)
    float* ws = (float*)d_ws;
    float* nx       = ws;            // normed tensor (reused for n1/n2/n3)
    float* conv_t   = ws + 1 * kT;   // conv output, (m, w, f)
    float* pw       = ws + 2 * kT;   // pointwise GEMM out, (m, f, w)
    float* qh       = ws + 3 * kT;   // q heads (m,nh,w,d)
    float* kh       = ws + 4 * kT;   // k heads (m,nh,w,d)
    float* vh_t     = ws + 5 * kT;   // v heads (m,nh,d,w)
    float* merged_t = ws + 6 * kT;   // attn out merged, (m, w, f)
    float* opw      = ws + 7 * kT;   // ol_pw out, (m, f, w)
    float* omix     = ws + 8 * kT;
    float* z        = ws + 9 * kT;
    float* hbuf     = ws + 10 * kT;
    float* ffn      = ws + 11 * kT;  // 4*kT (32 channels)
    float* probs    = ws + 15 * kT;  // kWW

    const unsigned lnGrid = (unsigned)(kC * kW);

    // --- block 1: self attention -------------------------------------------
    ft_layernorm<<<lnGrid, 256, 0, stream>>>(X, n1_w, n1_b, nx);
    run_attention(nx, nx, A1, M1, out_qk1, z,
                  conv_t, pw, qh, kh, vh_t, merged_t, opw, omix, probs, stream);
    ft_add<<<blocksFor(kT, 256), 256, 0, stream>>>(X, z, hbuf, kT);   // h = x + z

    // --- block 2: cross attention with skip (note: norm of z, not h) ------
    ft_layernorm<<<lnGrid, 256, 0, stream>>>(z, n2_w, n2_b, nx);
    run_attention(nx, SKIP, A2, M2, out_qk2, z,
                  conv_t, pw, qh, kh, vh_t, merged_t, opw, omix, probs, stream);
    ft_add<<<blocksFor(kT, 256), 256, 0, stream>>>(hbuf, z, hbuf, kT); // h += z2

    // --- FFN ----------------------------------------------------------------
    ft_layernorm<<<lnGrid, 256, 0, stream>>>(hbuf, n3_w, n3_b, nx);
    ft_conv3x3<<<blocksFor(4 * kT, 256), 256, 0, stream>>>(nx, kC, nullptr, 0, c1_w, c1_b, ffn, 4 * kC, 0);
    ft_gelu<<<blocksFor(4 * kT, 256), 256, 0, stream>>>(ffn, 4 * kT);
    ft_conv3x3<<<blocksFor(kT, 256), 256, 0, stream>>>(ffn, 4 * kC, nullptr, 0, c2_w, c2_b, z, kC, 0);
    ft_add<<<blocksFor(kT, 256), 256, 0, stream>>>(hbuf, z, out_h, kT); // h = h + f
}
